// Net_31198642438673
// MI455X (gfx1250) — matrix operand order repaired and checked by reading, not compiled, not run on hardware
//
#include <hip/hip_runtime.h>

typedef __attribute__((ext_vector_type(16))) _Float16 v16h;
typedef __attribute__((ext_vector_type(8)))  float    v8f;

constexpr int BB    = 8;
constexpr int NN    = 4096;
constexpr int KNB   = 20;                 // neighbors
constexpr int NPTS  = BB * NN;            // 32768 points
constexpr int NEDGE = NPTS * KNB;         // 655360 edges
constexpr int ETILES = NEDGE / 16;        // 40960 (exact)

typedef union { v16h v; _Float16 h[16]; } Frag;

// Order-preserving float -> uint encoding (for signed-value atomic max).
__device__ __forceinline__ unsigned fenc(float f) {
  int i = __float_as_int(f);
  return (unsigned)i ^ (((unsigned)(i >> 31)) | 0x80000000u);
}

// ---------------------------------------------------------------- utilities
__global__ void fill_u32(unsigned* __restrict__ p, unsigned val, int n) {
  int i = blockIdx.x * blockDim.x + threadIdx.x;
  if (i < n) p[i] = val;
}

__global__ void build_x0(const float* __restrict__ pos,
                         const float* __restrict__ feat,
                         float* __restrict__ x0) {
  int i = blockIdx.x * blockDim.x + threadIdx.x;
  if (i >= NPTS) return;
  x0[i * 4 + 0] = pos[i * 3 + 0];
  x0[i * 4 + 1] = pos[i * 3 + 1];
  x0[i * 4 + 2] = pos[i * 3 + 2];
  x0[i * 4 + 3] = feat[i];
}

// Convert weights to f16 row-major, fold Linear bias + BN into (scale, shift).
__global__ void pack_params(
    const float* __restrict__ W1, const float* __restrict__ b1, const float* __restrict__ g1,
    const float* __restrict__ be1, const float* __restrict__ mu1, const float* __restrict__ va1,
    const float* __restrict__ W2, const float* __restrict__ b2, const float* __restrict__ g2,
    const float* __restrict__ be2, const float* __restrict__ mu2, const float* __restrict__ va2,
    const float* __restrict__ W3, const float* __restrict__ b3, const float* __restrict__ g3,
    const float* __restrict__ be3, const float* __restrict__ mu3, const float* __restrict__ va3,
    const float* __restrict__ Wc, const float* __restrict__ bc, const float* __restrict__ gc,
    const float* __restrict__ bec, const float* __restrict__ muc, const float* __restrict__ vac,
    const float* __restrict__ Wl,
    _Float16* __restrict__ W1h, _Float16* __restrict__ W2h, _Float16* __restrict__ W3h,
    _Float16* __restrict__ Wch, _Float16* __restrict__ Wlh,
    float* __restrict__ s1, float* __restrict__ t1, float* __restrict__ s2, float* __restrict__ t2,
    float* __restrict__ s3, float* __restrict__ t3, float* __restrict__ sc, float* __restrict__ tc)
{
  int id = blockIdx.x * blockDim.x + threadIdx.x;
  if (id < 32 * 64) {             // W1 [8,64] zero-padded to [32,64] (K pad for WMMA)
    int r = id >> 6, c = id & 63;
    W1h[id] = (_Float16)((r < 8) ? W1[r * 64 + c] : 0.0f);
  }
  if (id < 64 * 64) { W2h[id] = (_Float16)W2[id]; W3h[id] = (_Float16)W3[id]; }
  if (id < 128 * 128) Wch[id] = (_Float16)Wc[id];
  if (id < 192 * 1024) Wlh[id] = (_Float16)Wl[id];
  if (id < 64) {
    float s;
    s = g1[id] * rsqrtf(va1[id] + 1e-5f); s1[id] = s; t1[id] = (b1[id] - mu1[id]) * s + be1[id];
    s = g2[id] * rsqrtf(va2[id] + 1e-5f); s2[id] = s; t2[id] = (b2[id] - mu2[id]) * s + be2[id];
    s = g3[id] * rsqrtf(va3[id] + 1e-5f); s3[id] = s; t3[id] = (b3[id] - mu3[id]) * s + be3[id];
  }
  if (id < 128) {
    float s = gc[id] * rsqrtf(vac[id] + 1e-5f);
    sc[id] = s; tc[id] = (bc[id] - muc[id]) * s + bec[id];
  }
}

// ------------------------------------------------------------------- knn
// LDS-tiled brute-force top-k: one block = 128 queries of a single batch
// (NN % 128 == 0); candidate rows are staged in LDS so all 128 queries share
// each global read (128x VMEM traffic reduction), and the next tile is
// software-prefetched (global_prefetch_b8) while the current one is scanned.
template <int D>
__global__ void __launch_bounds__(128) knn_kernel(const float* __restrict__ x,
                                                  int* __restrict__ idx) {
  constexpr int TILE = (D >= 64) ? 64 : 128;
  __shared__ float tile[TILE * D];
  const int t = threadIdx.x;
  const int q = blockIdx.x * 128 + t;
  const int b = q / NN;
  const int qi = q - b * NN;
  const float* xb = x + (size_t)b * NN * D;

  float4 qv[D / 4];
#pragma unroll
  for (int u = 0; u < D / 4; ++u)
    qv[u] = *(const float4*)(xb + (size_t)qi * D + 4 * u);

  float bd[KNB];
  int bi[KNB];
#pragma unroll
  for (int s = 0; s < KNB; ++s) { bd[s] = 3.4e38f; bi[s] = 0; }
  float worst = 3.4e38f;

  for (int j0 = 0; j0 < NN; j0 += TILE) {
    for (int u = t; u < TILE * D; u += 128)
      tile[u] = xb[(size_t)j0 * D + u];
    if (j0 + TILE < NN && t * 32 < TILE * D)   // one 128B line per thread
      __builtin_prefetch(xb + (size_t)(j0 + TILE) * D + t * 32, 0, 1);
    __syncthreads();

    for (int jj = 0; jj < TILE; ++jj) {
      float dist = 0.0f;
#pragma unroll
      for (int u = 0; u < D / 4; ++u) {
        float4 c4 = *(const float4*)(tile + jj * D + 4 * u);
        float dx = qv[u].x - c4.x, dy = qv[u].y - c4.y;
        float dz = qv[u].z - c4.z, dw = qv[u].w - c4.w;
        dist = fmaf(dx, dx, dist); dist = fmaf(dy, dy, dist);
        dist = fmaf(dz, dz, dist); dist = fmaf(dw, dw, dist);
      }
      if (dist < worst) {
        int p = KNB - 1;
        while (p > 0 && bd[p - 1] > dist) { bd[p] = bd[p - 1]; bi[p] = bi[p - 1]; --p; }
        bd[p] = dist; bi[p] = j0 + jj;
        worst = bd[KNB - 1];
      }
    }
    __syncthreads();
  }
  for (int s = 0; s < KNB; ++s) idx[(size_t)q * KNB + s] = bi[s];
}

// ------------------------------------------------- EdgeConv1: MLP(8->64->64->64)+max
__global__ void __launch_bounds__(32) conv1_kernel(
    const float* __restrict__ x0, const int* __restrict__ idx,
    const _Float16* __restrict__ W1h, const _Float16* __restrict__ W2h,
    const _Float16* __restrict__ W3h,
    const float* __restrict__ s1, const float* __restrict__ t1,
    const float* __restrict__ s2, const float* __restrict__ t2,
    const float* __restrict__ s3, const float* __restrict__ t3,
    int* __restrict__ x1bits)
{
  __shared__ float actA[16 * 64];
  __shared__ float actB[16 * 64];
  const int lane = threadIdx.x;
  const int m = lane & 15, hi = lane >> 4;
  const int e0 = blockIdx.x * 16;

  // A fragment for layer 1: msg = [xi, xj-xi], K=8 real, zero-padded to 32.
  // A(m,k): lane = m + 16*((k%16)>=8); slot = (k/16)*8 + (k%8)
  Frag a1;
#pragma unroll
  for (int p = 0; p < 16; ++p) a1.h[p] = (_Float16)0.0f;
  if (hi == 0) {                       // lanes 0-15 hold k=0..7 in slots 0..7
    int e = e0 + m;
    int pt = e / KNB;
    int b = pt / NN;
    int j = idx[e];
    const float* xi = x0 + (size_t)pt * 4;
    const float* xj = x0 + ((size_t)b * NN + j) * 4;
#pragma unroll
    for (int d = 0; d < 4; ++d) {
      float vi = xi[d];
      a1.h[d]     = (_Float16)vi;
      a1.h[4 + d] = (_Float16)(xj[d] - vi);
    }
  }
  // Layer 1: [16x32] x [32x64]
#pragma unroll
  for (int nt = 0; nt < 4; ++nt) {
    Frag bf;
    const _Float16* wr = W1h + nt * 16 + m;    // B(k,n): lane=n+16*h, slot=table-order k
#pragma unroll
    for (int n = 0; n < 16; ++n) bf.h[n] = wr[((n >> 3) * 16 + hi * 8 + (n & 7)) * 64];
    v8f c = {};
    c = __builtin_amdgcn_wmma_f32_16x16x32_f16(false, a1.v, false, bf.v, (short)0, c, false, false);
    int ch = nt * 16 + m;
    float scv = s1[ch], shv = t1[ch];
#pragma unroll
    for (int v = 0; v < 8; ++v)
      actA[(hi * 8 + v) * 64 + ch] = fmaxf(c[v] * scv + shv, 0.0f);
  }
  __syncthreads();

  // Layer 2: K=64 -> 2 chunks
  Frag a2[2];
#pragma unroll
  for (int ck = 0; ck < 2; ++ck)
#pragma unroll
    for (int p = 0; p < 16; ++p) {
      int k = ck * 32 + (p >> 3) * 16 + hi * 8 + (p & 7);
      a2[ck].h[p] = (_Float16)actA[m * 64 + k];
    }
#pragma unroll
  for (int nt = 0; nt < 4; ++nt) {
    v8f c = {};
#pragma unroll
    for (int ck = 0; ck < 2; ++ck) {
      Frag bf;
      const _Float16* wr = W2h + (ck * 32) * 64 + nt * 16 + m;
#pragma unroll
      for (int n = 0; n < 16; ++n) bf.h[n] = wr[((n >> 3) * 16 + hi * 8 + (n & 7)) * 64];
      c = __builtin_amdgcn_wmma_f32_16x16x32_f16(false, a2[ck].v, false, bf.v, (short)0, c, false, false);
    }
    int ch = nt * 16 + m;
    float scv = s2[ch], shv = t2[ch];
#pragma unroll
    for (int v = 0; v < 8; ++v)
      actB[(hi * 8 + v) * 64 + ch] = fmaxf(c[v] * scv + shv, 0.0f);
  }
  __syncthreads();

  // Layer 3 + max-aggregation over neighbor runs (post-ReLU >= 0 -> int atomic max)
  Frag a3[2];
#pragma unroll
  for (int ck = 0; ck < 2; ++ck)
#pragma unroll
    for (int p = 0; p < 16; ++p) {
      int k = ck * 32 + (p >> 3) * 16 + hi * 8 + (p & 7);
      a3[ck].h[p] = (_Float16)actB[m * 64 + k];
    }
#pragma unroll
  for (int nt = 0; nt < 4; ++nt) {
    v8f c = {};
#pragma unroll
    for (int ck = 0; ck < 2; ++ck) {
      Frag bf;
      const _Float16* wr = W3h + (ck * 32) * 64 + nt * 16 + m;
#pragma unroll
      for (int n = 0; n < 16; ++n) bf.h[n] = wr[((n >> 3) * 16 + hi * 8 + (n & 7)) * 64];
      c = __builtin_amdgcn_wmma_f32_16x16x32_f16(false, a3[ck].v, false, bf.v, (short)0, c, false, false);
    }
    int ch = nt * 16 + m;
    float scv = s3[ch], shv = t3[ch];
    int curpt = -1;
    float run = 0.0f;
#pragma unroll
    for (int v = 0; v < 8; ++v) {
      int e = e0 + hi * 8 + v;
      int pt = e / KNB;
      float y = fmaxf(c[v] * scv + shv, 0.0f);
      if (pt != curpt) {
        if (curpt >= 0) atomicMax(&x1bits[(size_t)curpt * 64 + ch], __float_as_int(run));
        curpt = pt; run = y;
      } else {
        run = fmaxf(run, y);
      }
    }
    atomicMax(&x1bits[(size_t)curpt * 64 + ch], __float_as_int(run));
  }
}

// ------------------------------------------------- EdgeConv2: MLP(128->128)+max
__global__ void __launch_bounds__(32) conv2_kernel(
    const float* __restrict__ x1, const int* __restrict__ idx,
    const _Float16* __restrict__ Wch,
    const float* __restrict__ sc2, const float* __restrict__ tc2,
    int* __restrict__ x2bits)
{
  __shared__ float msg[16 * 128];
  const int lane = threadIdx.x;
  const int m = lane & 15, hi = lane >> 4;
  const int e0 = blockIdx.x * 16;
  {
    int er = lane >> 1;            // edge row 0..15, two lanes per edge
    int hf = lane & 1;             // 0: xi half, 1: (xj - xi) half
    int e = e0 + er;
    int pt = e / KNB;
    int b = pt / NN;
    int j = idx[e];
    const float4* xi = (const float4*)(x1 + (size_t)pt * 64);
    const float4* xj = (const float4*)(x1 + ((size_t)b * NN + j) * 64);
    float* dst = msg + er * 128 + hf * 64;
    if (hf == 0) {
#pragma unroll
      for (int t = 0; t < 16; ++t) {
        float4 a = xi[t];
        dst[t * 4 + 0] = a.x; dst[t * 4 + 1] = a.y; dst[t * 4 + 2] = a.z; dst[t * 4 + 3] = a.w;
      }
    } else {
#pragma unroll
      for (int t = 0; t < 16; ++t) {
        float4 a = xi[t], bv = xj[t];
        dst[t * 4 + 0] = bv.x - a.x; dst[t * 4 + 1] = bv.y - a.y;
        dst[t * 4 + 2] = bv.z - a.z; dst[t * 4 + 3] = bv.w - a.w;
      }
    }
  }
  __syncthreads();

  Frag a[4];                        // K=128 -> 4 chunks
#pragma unroll
  for (int ck = 0; ck < 4; ++ck)
#pragma unroll
    for (int p = 0; p < 16; ++p) {
      int k = ck * 32 + (p >> 3) * 16 + hi * 8 + (p & 7);
      a[ck].h[p] = (_Float16)msg[m * 128 + k];
    }
#pragma unroll
  for (int nt = 0; nt < 8; ++nt) {
    v8f c = {};
#pragma unroll
    for (int ck = 0; ck < 4; ++ck) {
      Frag bf;
      const _Float16* wr = Wch + (ck * 32) * 128 + nt * 16 + m;
#pragma unroll
      for (int n = 0; n < 16; ++n) bf.h[n] = wr[((n >> 3) * 16 + hi * 8 + (n & 7)) * 128];
      c = __builtin_amdgcn_wmma_f32_16x16x32_f16(false, a[ck].v, false, bf.v, (short)0, c, false, false);
    }
    int ch = nt * 16 + m;
    float scv = sc2[ch], shv = tc2[ch];
    int curpt = -1;
    float run = 0.0f;
#pragma unroll
    for (int v = 0; v < 8; ++v) {
      int e = e0 + hi * 8 + v;
      int pt = e / KNB;
      float y = fmaxf(c[v] * scv + shv, 0.0f);
      if (pt != curpt) {
        if (curpt >= 0) atomicMax(&x2bits[(size_t)curpt * 128 + ch], __float_as_int(run));
        curpt = pt; run = y;
      } else {
        run = fmaxf(run, y);
      }
    }
    atomicMax(&x2bits[(size_t)curpt * 128 + ch], __float_as_int(run));
  }
}

// --------------------------------------- lin1 (192->1024) fused with global max pool
__global__ void __launch_bounds__(32) lin1max_kernel(
    const float* __restrict__ x1, const float* __restrict__ x2,
    const _Float16* __restrict__ Wlh, const float* __restrict__ bias,
    unsigned* __restrict__ out2enc)
{
  __shared__ float rows[16 * 192];
  const int lane = threadIdx.x;
  const int m = lane & 15, hi = lane >> 4;
  const int p0 = blockIdx.x * 16;       // 16 points, never straddles a batch (N%16==0)
  const int b = p0 / NN;
  const int cg = blockIdx.y;            // 128-column group

  for (int r = 0; r < 16; ++r) {
#pragma unroll
    for (int t = 0; t < 6; ++t) {
      int c = lane + 32 * t;
      float v = (c < 64) ? x1[(size_t)(p0 + r) * 64 + c]
                         : x2[(size_t)(p0 + r) * 128 + (c - 64)];
      rows[r * 192 + c] = v;
    }
  }
  __syncthreads();

  Frag a[6];                            // K=192 -> 6 chunks
#pragma unroll
  for (int ck = 0; ck < 6; ++ck)
#pragma unroll
    for (int p = 0; p < 16; ++p) {
      int k = ck * 32 + (p >> 3) * 16 + hi * 8 + (p & 7);
      a[ck].h[p] = (_Float16)rows[m * 192 + k];
    }
  for (int nt = 0; nt < 8; ++nt) {
    int colbase = cg * 128 + nt * 16;
    v8f c = {};
#pragma unroll
    for (int ck = 0; ck < 6; ++ck) {
      Frag bf;
      const _Float16* wr = Wlh + (size_t)(ck * 32) * 1024 + colbase + m;
#pragma unroll
      for (int n = 0; n < 16; ++n) bf.h[n] = wr[(size_t)((n >> 3) * 16 + hi * 8 + (n & 7)) * 1024];
      c = __builtin_amdgcn_wmma_f32_16x16x32_f16(false, a[ck].v, false, bf.v, (short)0, c, false, false);
    }
    int ch = colbase + m;
    float bv = bias[ch];
    float loc = -3.4e38f;
#pragma unroll
    for (int v = 0; v < 8; ++v) loc = fmaxf(loc, c[v] + bv);
    atomicMax(&out2enc[b * 1024 + ch], fenc(loc));
  }
}

// -------------------------------------------------- head MLP + log_softmax
__global__ void __launch_bounds__(256) head_kernel(
    const unsigned* __restrict__ out2enc,
    const float* __restrict__ W0, const float* __restrict__ b0,
    const float* __restrict__ W1, const float* __restrict__ b1,
    const float* __restrict__ W2, const float* __restrict__ b2,
    float* __restrict__ out)
{
  __shared__ float xb[1024];
  __shared__ float h1[512];
  __shared__ float h2[256];
  __shared__ float lg[10];
  const int b = blockIdx.x, t = threadIdx.x;
  for (int c = t; c < 1024; c += 256) {
    unsigned u = out2enc[b * 1024 + c];
    int i = (u & 0x80000000u) ? (int)(u ^ 0x80000000u) : ~(int)u;
    xb[c] = __int_as_float(i);
  }
  __syncthreads();
  for (int o = t; o < 512; o += 256) {
    float acc = b0[o];
    for (int k = 0; k < 1024; ++k) acc = fmaf(xb[k], W0[k * 512 + o], acc);
    h1[o] = fmaxf(acc, 0.0f);
  }
  __syncthreads();
  {
    float acc = b1[t];
    for (int k = 0; k < 512; ++k) acc = fmaf(h1[k], W1[k * 256 + t], acc);
    h2[t] = fmaxf(acc, 0.0f);
  }
  __syncthreads();
  if (t < 10) {
    float acc = b2[t];
    for (int k = 0; k < 256; ++k) acc = fmaf(h2[k], W2[k * 10 + t], acc);
    lg[t] = acc;
  }
  __syncthreads();
  if (t == 0) {
    float mx = lg[0];
    for (int i = 1; i < 10; ++i) mx = fmaxf(mx, lg[i]);
    float s = 0.0f;
    for (int i = 0; i < 10; ++i) s += expf(lg[i] - mx);
    float lse = mx + logf(s);
    for (int i = 0; i < 10; ++i) out[b * 10 + i] = lg[i] - lse;
  }
}

// ---------------------------------------------------------------- launcher
extern "C" void kernel_launch(void* const* d_in, const int* in_sizes, int n_in,
                              void* d_out, int out_size, void* d_ws, size_t ws_size,
                              hipStream_t stream) {
  (void)in_sizes; (void)n_in; (void)out_size; (void)ws_size;
  const float* pos  = (const float*)d_in[0];
  const float* feat = (const float*)d_in[1];
  // d_in[2] = k (compile-time constant 20)
  const float* c1W[3]  = {(const float*)d_in[3],  (const float*)d_in[9],  (const float*)d_in[15]};
  const float* c1b[3]  = {(const float*)d_in[4],  (const float*)d_in[10], (const float*)d_in[16]};
  const float* c1g[3]  = {(const float*)d_in[5],  (const float*)d_in[11], (const float*)d_in[17]};
  const float* c1be[3] = {(const float*)d_in[6],  (const float*)d_in[12], (const float*)d_in[18]};
  const float* c1mu[3] = {(const float*)d_in[7],  (const float*)d_in[13], (const float*)d_in[19]};
  const float* c1va[3] = {(const float*)d_in[8],  (const float*)d_in[14], (const float*)d_in[20]};
  const float* WcF  = (const float*)d_in[21];
  const float* bcF  = (const float*)d_in[22];
  const float* gcF  = (const float*)d_in[23];
  const float* becF = (const float*)d_in[24];
  const float* mucF = (const float*)d_in[25];
  const float* vacF = (const float*)d_in[26];
  const float* WlF  = (const float*)d_in[27];
  const float* blF  = (const float*)d_in[28];
  const float* hW0  = (const float*)d_in[29];
  const float* hb0  = (const float*)d_in[30];
  const float* hW1  = (const float*)d_in[31];
  const float* hb1  = (const float*)d_in[32];
  const float* hW2  = (const float*)d_in[33];
  const float* hb2  = (const float*)d_in[34];

  size_t off = 0;
  auto take = [&](size_t bytes) -> void* {
    void* p = (char*)d_ws + off;
    off += (bytes + 255) & ~(size_t)255;
    return p;
  };
  float*    x0      = (float*)take((size_t)NPTS * 4 * sizeof(float));
  int*      idx1    = (int*)take((size_t)NEDGE * sizeof(int));
  float*    x1      = (float*)take((size_t)NPTS * 64 * sizeof(float));
  int*      idx2    = (int*)take((size_t)NEDGE * sizeof(int));
  float*    x2      = (float*)take((size_t)NPTS * 128 * sizeof(float));
  unsigned* out2enc = (unsigned*)take((size_t)BB * 1024 * sizeof(unsigned));
  _Float16* W1h = (_Float16*)take(32 * 64 * 2);
  _Float16* W2h = (_Float16*)take(64 * 64 * 2);
  _Float16* W3h = (_Float16*)take(64 * 64 * 2);
  _Float16* Wch = (_Float16*)take(128 * 128 * 2);
  _Float16* Wlh = (_Float16*)take(192 * 1024 * 2);
  float* s1 = (float*)take(64 * 4);  float* t1 = (float*)take(64 * 4);
  float* s2 = (float*)take(64 * 4);  float* t2 = (float*)take(64 * 4);
  float* s3 = (float*)take(64 * 4);  float* t3 = (float*)take(64 * 4);
  float* sc = (float*)take(128 * 4); float* tc = (float*)take(128 * 4);

  pack_params<<<dim3(768), dim3(256), 0, stream>>>(
      c1W[0], c1b[0], c1g[0], c1be[0], c1mu[0], c1va[0],
      c1W[1], c1b[1], c1g[1], c1be[1], c1mu[1], c1va[1],
      c1W[2], c1b[2], c1g[2], c1be[2], c1mu[2], c1va[2],
      WcF, bcF, gcF, becF, mucF, vacF, WlF,
      W1h, W2h, W3h, Wch, Wlh, s1, t1, s2, t2, s3, t3, sc, tc);
  build_x0<<<dim3(NPTS / 256), dim3(256), 0, stream>>>(pos, feat, x0);
  knn_kernel<4><<<dim3(NPTS / 128), dim3(128), 0, stream>>>(x0, idx1);
  fill_u32<<<dim3((NPTS * 64 + 255) / 256), dim3(256), 0, stream>>>((unsigned*)x1, 0u, NPTS * 64);
  conv1_kernel<<<dim3(ETILES), dim3(32), 0, stream>>>(x0, idx1, W1h, W2h, W3h,
                                                      s1, t1, s2, t2, s3, t3, (int*)x1);
  knn_kernel<64><<<dim3(NPTS / 128), dim3(128), 0, stream>>>(x1, idx2);
  fill_u32<<<dim3((NPTS * 128 + 255) / 256), dim3(256), 0, stream>>>((unsigned*)x2, 0u, NPTS * 128);
  conv2_kernel<<<dim3(ETILES), dim3(32), 0, stream>>>(x1, idx2, Wch, sc, tc, (int*)x2);
  fill_u32<<<dim3((BB * 1024 + 255) / 256), dim3(256), 0, stream>>>(out2enc, 0u, BB * 1024);
  lin1max_kernel<<<dim3(NPTS / 16, 8), dim3(32), 0, stream>>>(x1, x2, Wlh, blF, out2enc);
  head_kernel<<<dim3(BB), dim3(256), 0, stream>>>(out2enc, hW0, hb0, hW1, hb1, hW2, hb2,
                                                  (float*)d_out);
}
